// GGMLQuantizedLinear_1855425872551
// MI455X (gfx1250) — matrix operand order repaired and checked by reading, not compiled, not run on hardware
//
#include <hip/hip_runtime.h>

// ---- types (wave32, gfx1250) ----
typedef _Float16 h2   __attribute__((ext_vector_type(2)));
typedef _Float16 v8h  __attribute__((ext_vector_type(8)));
typedef _Float16 v16h __attribute__((ext_vector_type(16)));
typedef float    v8f  __attribute__((ext_vector_type(8)));
typedef float    v4f  __attribute__((ext_vector_type(4)));
typedef int      v4i  __attribute__((ext_vector_type(4)));

// ---- problem constants (B=2,S=2048 -> M=4096; O=N=4096; K=4096) ----
#define M_TOT   4096
#define N_TOT   4096
#define K_TOT   4096
#define GROUPSZ 32
#define NGROUPS (K_TOT / GROUPSZ)

// ---- tiling ----
#define BM  128         // M rows per block
#define BN  128         // N cols per block
#define KB  32          // K per iteration == quant group size
#define NK  (K_TOT / KB)
#define LDA 40          // LDS row pitch in halfs (32 + 8 pad; rows stay 16B aligned)
#define LDB 40

__device__ __forceinline__ v16h cat16(v8h lo, v8h hi) {
    return __builtin_shufflevector(lo, hi,
        0, 1, 2, 3, 4, 5, 6, 7, 8, 9, 10, 11, 12, 13, 14, 15);
}

__device__ __forceinline__ v8f wmma16(v16h a, v16h b, v8f c) {
    return __builtin_amdgcn_wmma_f32_16x16x32_f16(
        false, a, false, b, (short)0, c, false, false);
}

struct Raw {            // one K-chunk of per-thread staging data, held in VGPRs
    v4f x0, x1, x2, x3; // 16 fp32 of x
    v4i w0, w1;         // 8 packed weight bytes (16 nibbles)
    float sf, zf;       // group scale / zero-point
};

__global__ __launch_bounds__(256) void q4_linear_wmma(
    const float* __restrict__ x,        // [M_TOT, K_TOT]
    const int*   __restrict__ wpacked,  // [N_TOT, K_TOT/2] one byte per int32
    const float* __restrict__ scales,   // [N_TOT, NGROUPS]
    const float* __restrict__ zps,      // [N_TOT, NGROUPS]
    const float* __restrict__ bias,     // [N_TOT]
    float* __restrict__ out)            // [M_TOT, N_TOT]
{
    // double-buffered staging tiles (40 KB total; WGP has 320 KB)
    __shared__ _Float16 lds_a[2][BM * LDA];
    __shared__ _Float16 lds_b[2][BN * LDB];

    const int tid  = threadIdx.x;
    const int lane = tid & 31;
    const int wave = tid >> 5;
    const int wr   = wave >> 1;   // 0..3 -> wave M offset (x32)
    const int wc   = wave & 1;    // 0..1 -> wave N offset (x64)

    const int blockN = blockIdx.x * BN;
    const int blockM = blockIdx.y * BM;

    // ---- staging roles: each thread owns one row-half (16 elements) of A and B ----
    const int ar  = tid >> 1;           // 0..127
    const int ah8 = (tid & 1) * 16;     // half-offset within 32-wide row
    const float* xrow    = x + (size_t)(blockM + ar) * K_TOT + ah8;
    const int    aoff_st = ar * LDA + ah8;

    const int br  = tid >> 1;           // 0..127
    const int bh8 = (tid & 1) * 16;     // half-offset (16 halfs == 8 packed int32)
    const int*   wrow = wpacked + (size_t)(blockN + br) * (K_TOT / 2) + (bh8 >> 1);
    const float* srow = scales  + (size_t)(blockN + br) * NGROUPS;
    const float* zrow = zps     + (size_t)(blockN + br) * NGROUPS;
    const int    boff_st = br * LDB + bh8;

    // ---- per-lane fragment offsets (documented gfx1250 wave32 layouts) ----
    // A 16x32 f16: lane<16 -> K {0..7,16..23}; lane>=16 -> K {8..15,24..31}
    const int mf    = wr * 32 + (lane & 15);
    const int abase = (lane < 16) ? 0 : 8;
    const int afr0  = mf * LDA + abase;
    const int afr1  = (mf + 16) * LDA + abase;
    // B 32x16 f16: lane<16 -> K {0..7,16..23}; lane>=16 -> K {8..15,24..31}
    const int nf    = wc * 64 + (lane & 15);
    const int bbase = (lane < 16) ? 0 : 8;
    int bfr[4];
    #pragma unroll
    for (int t = 0; t < 4; ++t) bfr[t] = (nf + t * 16) * LDB + bbase;

    v8f acc0[4] = {{}, {}, {}, {}};   // rows wr*32 + 0..15
    v8f acc1[4] = {{}, {}, {}, {}};   // rows wr*32 + 16..31

    const h2 k1024 = {(_Float16)1024.0f, (_Float16)1024.0f};

    // ---- global-load one K-chunk into registers ----
    auto ld = [&](int kb) {
        Raw r;
        const float* xp = xrow + (size_t)kb * KB;
        r.x0 = *(const v4f*)(xp);
        r.x1 = *(const v4f*)(xp + 4);
        r.x2 = *(const v4f*)(xp + 8);
        r.x3 = *(const v4f*)(xp + 12);
        const int* wp = wrow + (size_t)kb * (KB / 2);
        r.w0 = *(const v4i*)(wp);
        r.w1 = *(const v4i*)(wp + 4);
        r.sf = srow[kb];
        r.zf = zrow[kb];
        return r;
    };

    // ---- convert + store one K-chunk into LDS buffer bi ----
    auto st = [&](const Raw& r, int bi) {
        // A: 16 fp32 -> 16 f16
        v8h ha, hb;
        #pragma unroll
        for (int i = 0; i < 4; ++i) {
            ha[i]     = (_Float16)r.x0[i];
            ha[4 + i] = (_Float16)r.x1[i];
            hb[i]     = (_Float16)r.x2[i];
            hb[4 + i] = (_Float16)r.x3[i];
        }
        *(v8h*)&lds_a[bi][aoff_st]     = ha;
        *(v8h*)&lds_a[bi][aoff_st + 8] = hb;

        // B: 16 nibbles -> 16 f16 via magic-exponent trick, all packed-f16 ops
        const _Float16 sh = (_Float16)r.sf;
        const _Float16 ch = (_Float16)(-r.zf * r.sf);
        const h2 s2 = {sh, sh};
        const h2 c2 = {ch, ch};
        v4i p0, p1;
        #pragma unroll
        for (int j = 0; j < 4; ++j) {
            {
                unsigned b = (unsigned)r.w0[j];
                unsigned t = ((b & 15u) | ((b << 12) & 0x000F0000u)) | 0x64006400u;
                h2 q = __builtin_bit_cast(h2, t) - k1024;      // exact (q0,q1)
                p0[j] = __builtin_bit_cast(int, __builtin_elementwise_fma(q, s2, c2));
            }
            {
                unsigned b = (unsigned)r.w1[j];
                unsigned t = ((b & 15u) | ((b << 12) & 0x000F0000u)) | 0x64006400u;
                h2 q = __builtin_bit_cast(h2, t) - k1024;
                p1[j] = __builtin_bit_cast(int, __builtin_elementwise_fma(q, s2, c2));
            }
        }
        *(v4i*)&lds_b[bi][boff_st]     = p0;
        *(v4i*)&lds_b[bi][boff_st + 8] = p1;
    };

    // prologue: fill buffer 0
    {
        Raw r0 = ld(0);
        st(r0, 0);
    }
    __syncthreads();

    for (int kb = 0; kb < NK; ++kb) {
        const int bi    = kb & 1;
        const int knext = (kb + 1 < NK) ? kb + 1 : kb;  // last iter: harmless restage

        // 1) issue next-chunk global loads first (latency hidden behind WMMAs)
        Raw r = ld(knext);

        __builtin_prefetch(xrow + (size_t)(kb + 2) * KB, 0, 3);
        __builtin_prefetch(wrow + (size_t)(kb + 2) * (KB / 2), 0, 3);

        // 2) all fragment loads up-front, distinct registers (no per-WMMA dscnt stalls)
        const _Float16* A = lds_a[bi];
        const _Float16* B = lds_b[bi];
        v16h a0 = cat16(*(const v8h*)&A[afr0], *(const v8h*)&A[afr0 + 16]);
        v16h a1 = cat16(*(const v8h*)&A[afr1], *(const v8h*)&A[afr1 + 16]);
        v16h bt[4];
        #pragma unroll
        for (int t = 0; t < 4; ++t)
            bt[t] = cat16(*(const v8h*)&B[bfr[t]], *(const v8h*)&B[bfr[t] + 16]);

        // 3) 32x64 per-wave outer product: 8 independent WMMAs back-to-back
        #pragma unroll
        for (int t = 0; t < 4; ++t) {
            acc0[t] = wmma16(a0, bt[t], acc0[t]);
            acc1[t] = wmma16(a1, bt[t], acc1[t]);
        }

        // 4) dequant/convert (VALU co-executes with WMMAs) + stores to other buffer
        st(r, bi ^ 1);

        __syncthreads();
    }

    // ---- epilogue: C/D layout VGPR i -> (M=i, N=lane) / (M=i+8, N=lane-16) ----
    const int row0 = blockM + wr * 32 + ((lane < 16) ? 0 : 8);
    const int cb   = blockN + wc * 64 + (lane & 15);
    #pragma unroll
    for (int t = 0; t < 4; ++t) {
        const int c = cb + t * 16;
        const float bz = bias[c];
        #pragma unroll
        for (int i = 0; i < 8; ++i) {
            out[(size_t)(row0 + i) * N_TOT + c]      = acc0[t][i] + bz;
            out[(size_t)(row0 + 16 + i) * N_TOT + c] = acc1[t][i] + bz;
        }
    }
}

extern "C" void kernel_launch(void* const* d_in, const int* in_sizes, int n_in,
                              void* d_out, int out_size, void* d_ws, size_t ws_size,
                              hipStream_t stream) {
    (void)in_sizes; (void)n_in; (void)out_size; (void)d_ws; (void)ws_size;
    const float* x       = (const float*)d_in[0];
    const int*   wpacked = (const int*)  d_in[1];
    const float* scales  = (const float*)d_in[2];
    const float* zps     = (const float*)d_in[3];
    const float* bias    = (const float*)d_in[4];
    float* out = (float*)d_out;

    dim3 grid(N_TOT / BN, M_TOT / BM);
    dim3 block(256);
    q4_linear_wmma<<<grid, block, 0, stream>>>(x, wpacked, scales, zps, bias, out);
}
